// AE_45810121179173
// MI455X (gfx1250) — matrix operand order repaired and checked by reading, not compiled, not run on hardware
//
#include <hip/hip_runtime.h>
#include <hip/hip_bf16.h>
#include <math.h>

typedef __attribute__((ext_vector_type(16))) _Float16 v16h;
typedef __attribute__((ext_vector_type(8)))  _Float16 v8h;
typedef __attribute__((ext_vector_type(8)))  float    v8f;

#define WAVES 8
#define NB 8   // batch

// ---------------------------------------------------------------------------
// Spiral conv as gather+GEMM with WMMA f16->f32.
// in:   [NB, Vq, CIN] f32
// idx:  [Vq, 9] int32
// W:    [K=9*CIN, N] f32
// bias: [N] f32
// out:  [NB, Vq, N] f32
// Rows are flattened (b*Vq + v), M = NB*Vq. Each wave computes a 16 x (NT*16)
// output tile: A is gathered/staged ONCE per K-chunk and reused across all
// NT n-tiles (the gather is the bandwidth bottleneck).
// ---------------------------------------------------------------------------
template <int CIN, int NT>
__global__ __launch_bounds__(256) void spiral_conv_wmma(
    const float* __restrict__ in, const int* __restrict__ idx,
    const float* __restrict__ W, const float* __restrict__ bias,
    float* __restrict__ out,
    int Vq, int N, int M, int apply_elu)
{
    constexpr int K  = 9 * CIN;
    constexpr int NK = (K + 31) / 32;

    __shared__ alignas(16) _Float16 lds_w[NT][512];        // [nt][n][k]
    __shared__ alignas(16) _Float16 lds_a[WAVES][512];     // per-wave [m][k]

    const int tid  = threadIdx.x;
    const int wave = tid >> 5;
    const int lane = tid & 31;
    const int m0   = (blockIdx.x * WAVES + wave) * 16;

    v8f acc[NT];
    #pragma unroll
    for (int t = 0; t < NT; ++t) acc[t] = (v8f){};

    _Float16* la = &lds_a[wave][0];

    for (int kc = 0; kc < NK; ++kc) {
        const int k0 = kc << 5;
        __syncthreads();  // protect LDS reuse across chunks

        // Stage NT weight tiles, transposed to [n][k] (f32 -> f16).
        for (int i = tid; i < NT * 512; i += 256) {
            const int nt = i >> 9;
            const int n  = (i >> 5) & 15;
            const int k  = i & 31;
            const int gk = k0 + k, gn = nt * 16 + n;
            float w = 0.f;
            if (gk < K && gn < N) w = W[(size_t)gk * N + gn];
            lds_w[nt][n * 32 + k] = (_Float16)w;
        }

        // Stage this wave's A chunk [16 m][32 k] via the spiral gather.
        // m == j is lane-uniform -> b, v computed scalar per row.
        for (int j = 0; j < 16; ++j) {
            const int row = m0 + j;
            const int k   = lane;
            const int gk  = k0 + k;
            float a = 0.f;
            if (row < M && gk < K) {
                const int b  = row / Vq;           // lane-uniform (scalar)
                const int v  = row - b * Vq;
                const int l  = gk / CIN;           // compile-time strength-reduced
                const int ci = gk - l * CIN;
                const int src = idx[(size_t)v * 9 + l];
                a = in[((size_t)b * Vq + src) * CIN + ci];
            }
            la[j * 32 + k] = (_Float16)a;
        }
        __syncthreads();

        // Build fragments per CDNA5 16-bit layouts.
        const int r  = lane & 15;
        const int hi = lane >> 4;
        // A (16x32): lanes 0-15 hold K 0-7 & 16-23; lanes 16-31 hold K 8-15 & 24-31.
        const v8h* pa = (const v8h*)&la[r * 32];
        v8h a_lo = pa[hi];        // K = hi*8 .. hi*8+7
        v8h a_hi = pa[2 + hi];    // K = 16 + hi*8 ..
        v16h av = __builtin_shufflevector(a_lo, a_hi,
            0,1,2,3,4,5,6,7,8,9,10,11,12,13,14,15);

        // NT back-to-back WMMAs reusing the A fragment.
        #pragma unroll
        for (int nt = 0; nt < NT; ++nt) {
            // B (32x16): lanes 0-15 hold K 0-7 & 16-23; lanes 16-31 hold K 8-15 & 24-31.
            const v8h* pb = (const v8h*)&lds_w[nt][r * 32];
            v8h b_lo = pb[hi];
            v8h b_hi = pb[2 + hi];
            v16h bv = __builtin_shufflevector(b_lo, b_hi,
                0,1,2,3,4,5,6,7,8,9,10,11,12,13,14,15);
            acc[nt] = __builtin_amdgcn_wmma_f32_16x16x32_f16(
                /*neg_a=*/false, av, /*neg_b=*/false, bv,
                /*c_mod=*/(short)0, acc[nt], /*reuse_a=*/false, /*reuse_b=*/false);
        }
    }

    // Epilogue: C/D layout -> lane n = lane&15; VGPR r -> M = r + 8*(lane>=16).
    const int hi = lane >> 4;
    #pragma unroll
    for (int nt = 0; nt < NT; ++nt) {
        const int n = nt * 16 + (lane & 15);
        if (n < N) {
            const float bsv = bias[n];
            #pragma unroll
            for (int r8 = 0; r8 < 8; ++r8) {
                const int row = m0 + r8 + hi * 8;
                if (row < M) {
                    float v = acc[nt][r8] + bsv;
                    if (apply_elu) v = (v > 0.f) ? v : (expf(v) - 1.f);
                    out[(size_t)row * N + n] = v;
                }
            }
        }
    }
}

// ---------------------------------------------------------------------------
__global__ void zero_f32(float* __restrict__ p, size_t n) {
    size_t i = (size_t)blockIdx.x * blockDim.x + threadIdx.x;
    if (i < n) p[i] = 0.f;
}

// Sparse pool: out[b, row[e], c] += in[b, col[e], c] * val[e]
__global__ void pool_scatter(const float* __restrict__ in,
                             const int* __restrict__ row, const int* __restrict__ col,
                             const float* __restrict__ val,
                             float* __restrict__ out,
                             int nnz, int C, int Vin, int Vout)
{
    int i = blockIdx.x * blockDim.x + threadIdx.x;
    if (i >= nnz * C) return;
    const int e = i / C, c = i - e * C;
    const int r = row[e], cl = col[e];
    const float v = val[e];
    #pragma unroll
    for (int b = 0; b < NB; ++b) {
        const float g = in[((size_t)b * Vin + cl) * C + c] * v;
        atomicAdd(&out[((size_t)b * Vout + r) * C + c], g);
    }
}

// z[b,o] = sum_k h[b,k]*W[k,o] + bias[o]; one block per (b,o), K large.
__global__ __launch_bounds__(256) void dense_enc(
    const float* __restrict__ h, const float* __restrict__ W,
    const float* __restrict__ bias, float* __restrict__ z, int K, int N)
{
    const int o = blockIdx.x % N;
    const int b = blockIdx.x / N;
    float s = 0.f;
    for (int k = threadIdx.x; k < K; k += 256)
        s += h[(size_t)b * K + k] * W[(size_t)k * N + o];
    __shared__ float red[256];
    red[threadIdx.x] = s;
    __syncthreads();
    for (int off = 128; off > 0; off >>= 1) {
        if (threadIdx.x < off) red[threadIdx.x] += red[threadIdx.x + off];
        __syncthreads();
    }
    if (threadIdx.x == 0) z[(size_t)b * N + o] = red[0] + bias[o];
}

// out[b,j] = sum_k z[b,k]*W[k,j] + bias[j]; K small (256).
__global__ void dense_dec(const float* __restrict__ z, const float* __restrict__ W,
                          const float* __restrict__ bias, float* __restrict__ out,
                          int K, int N)
{
    int i = blockIdx.x * blockDim.x + threadIdx.x;
    if (i >= NB * N) return;
    const int b = i / N, j = i - b * N;
    float s = bias[j];
    for (int k = 0; k < K; ++k)
        s += z[(size_t)b * K + k] * W[(size_t)k * N + j];
    out[(size_t)b * N + j] = s;
}

// ---------------------------------------------------------------------------
extern "C" void kernel_launch(void* const* d_in, const int* in_sizes, int n_in,
                              void* d_out, int out_size, void* d_ws, size_t ws_size,
                              hipStream_t stream)
{
    (void)in_sizes; (void)n_in; (void)out_size; (void)ws_size;

    const float* x = (const float*)d_in[0];
    auto SP = [&](int l){ return (const int*)  d_in[1 + l*7 + 0]; };
    auto DR = [&](int l){ return (const int*)  d_in[1 + l*7 + 1]; };
    auto DC = [&](int l){ return (const int*)  d_in[1 + l*7 + 2]; };
    auto DV = [&](int l){ return (const float*)d_in[1 + l*7 + 3]; };
    auto UR = [&](int l){ return (const int*)  d_in[1 + l*7 + 4]; };
    auto UC = [&](int l){ return (const int*)  d_in[1 + l*7 + 5]; };
    auto UV = [&](int l){ return (const float*)d_in[1 + l*7 + 6]; };
    auto WE = [&](int l){ return (const float*)d_in[29 + 2*l]; };
    auto BE = [&](int l){ return (const float*)d_in[30 + 2*l]; };
    const float* Wenc = (const float*)d_in[37];
    const float* benc = (const float*)d_in[38];
    const float* Wdec = (const float*)d_in[39];
    const float* bdec = (const float*)d_in[40];
    auto WD = [&](int l){ return (const float*)d_in[41 + 2*l]; };
    auto BD = [&](int l){ return (const float*)d_in[42 + 2*l]; };
    const float* Wf  = (const float*)d_in[49];
    const float* bf_ = (const float*)d_in[50];

    // Workspace: two ping-pong buffers sized for [8, 40000, 32] f32, plus z.
    const size_t bufElems = (size_t)NB * 40000 * 32;
    float* bufA = (float*)d_ws;
    float* bufB = bufA + bufElems;
    float* zbuf = bufB + bufElems;

    auto conv = [&](const float* in, const int* idx, const float* W, const float* bias,
                    float* out, int Vq, int cin, int cout, bool elu) {
        const int N = cout, M = NB * Vq;
        const int elu_i = elu ? 1 : 0;
        dim3 grid((M + 16 * WAVES - 1) / (16 * WAVES));
        const int NT = (N + 15) / 16;
        if (cin == 3 && NT == 2)
            spiral_conv_wmma<3, 2><<<grid, 256, 0, stream>>>(in, idx, W, bias, out, Vq, N, M, elu_i);
        else if (cin == 32 && NT == 1)
            spiral_conv_wmma<32, 1><<<grid, 256, 0, stream>>>(in, idx, W, bias, out, Vq, N, M, elu_i);
        else if (cin == 32 && NT == 2)
            spiral_conv_wmma<32, 2><<<grid, 256, 0, stream>>>(in, idx, W, bias, out, Vq, N, M, elu_i);
        else if (cin == 32 && NT == 4)
            spiral_conv_wmma<32, 4><<<grid, 256, 0, stream>>>(in, idx, W, bias, out, Vq, N, M, elu_i);
        else if (cin == 64 && NT == 2)
            spiral_conv_wmma<64, 2><<<grid, 256, 0, stream>>>(in, idx, W, bias, out, Vq, N, M, elu_i);
        else if (cin == 64 && NT == 4)
            spiral_conv_wmma<64, 4><<<grid, 256, 0, stream>>>(in, idx, W, bias, out, Vq, N, M, elu_i);
    };
    auto poolop = [&](const float* in, const int* row, const int* col, const float* val,
                      float* out, int nnz, int C, int Vin, int Vout) {
        const size_t n = (size_t)NB * Vout * C;
        zero_f32<<<(unsigned)((n + 255) / 256), 256, 0, stream>>>(out, n);
        const int tot = nnz * C;
        pool_scatter<<<(tot + 255) / 256, 256, 0, stream>>>(in, row, col, val, out,
                                                            nnz, C, Vin, Vout);
    };

    // ---------------- Encoder ----------------
    conv(x,    SP(0), WE(0), BE(0), bufA, 40000,  3, 32, true);
    poolop(bufA, DR(0), DC(0), DV(0), bufB, 40000, 32, 40000, 10000);
    conv(bufB, SP(1), WE(1), BE(1), bufA, 10000, 32, 32, true);
    poolop(bufA, DR(1), DC(1), DV(1), bufB, 10000, 32, 10000, 2500);
    conv(bufB, SP(2), WE(2), BE(2), bufA, 2500,  32, 32, true);
    poolop(bufA, DR(2), DC(2), DV(2), bufB, 2500, 32, 2500, 625);
    conv(bufB, SP(3), WE(3), BE(3), bufA, 625,   32, 64, true);
    poolop(bufA, DR(3), DC(3), DV(3), bufB, 625, 64, 625, 160);

    // ---------------- Latent ----------------
    dense_enc<<<NB * 256, 256, 0, stream>>>(bufB, Wenc, benc, zbuf, 160 * 64, 256);
    {
        const int tot = NB * 160 * 64;
        dense_dec<<<(tot + 255) / 256, 256, 0, stream>>>(zbuf, Wdec, bdec, bufA,
                                                         256, 160 * 64);
    }

    // ---------------- Decoder ----------------
    poolop(bufA, UR(3), UC(3), UV(3), bufB, 625,   64, 160,   625);
    conv(bufB, SP(3), WD(3), BD(3), bufA, 625,   64, 64, true);
    poolop(bufA, UR(2), UC(2), UV(2), bufB, 2500,  64, 625,   2500);
    conv(bufB, SP(2), WD(2), BD(2), bufA, 2500,  64, 32, true);
    poolop(bufA, UR(1), UC(1), UV(1), bufB, 10000, 32, 2500,  10000);
    conv(bufB, SP(1), WD(1), BD(1), bufA, 10000, 32, 32, true);
    poolop(bufA, UR(0), UC(0), UV(0), bufB, 40000, 32, 10000, 40000);
    conv(bufB, SP(0), WD(0), BD(0), bufA, 40000, 32, 32, true);

    // Final projection to 3 channels, no ELU, straight into d_out.
    conv(bufA, SP(0), Wf, bf_, (float*)d_out, 40000, 32, 3, false);
}
